// GraphAttention_74199855006211
// MI455X (gfx1250) — matrix operand order repaired and checked by reading, not compiled, not run on hardware
//
#include <hip/hip_runtime.h>

// MI455X / gfx1250, wave32. Matrix math via v_wmma_f32_16x16x32_f16.
// Memory-bound (~26 GFLOP vs ~250 MB HBM @ 23.3 TB/s). All WMMA operands are
// fed by contiguous b128 loads:
//  - kernels pre-transposed once to WkT[h][c][k] f16 (0.5 MB, L2-resident) so
//    the projection GEMM reads BOTH operands straight from global memory:
//    no LDS, no barriers in the K-loop.
//  - feats stored transposed (featsT[h,b][c][n]) so the node GEMM B-fragment
//    is two global b128 loads; attn tiles live only in LDS (f16, 64 KB).

typedef __attribute__((ext_vector_type(16))) _Float16 v16h;
typedef __attribute__((ext_vector_type(8)))  _Float16 v8h;
typedef __attribute__((ext_vector_type(2)))  _Float16 v2h;
typedef __attribute__((ext_vector_type(8)))  float    v8f;

#define B_    128
#define N_    256
#define F_IN  512
#define H_    8
#define FD    64
#define NEG_BIG (-1.0e10f)
#define SLOPE 0.2f

static __device__ __forceinline__ v16h cat8(v8h lo, v8h hi) {
  return __builtin_shufflevector(lo, hi, 0, 1, 2, 3, 4, 5, 6, 7,
                                 8, 9, 10, 11, 12, 13, 14, 15);
}

// ---------------------------------------------------------------------------
// Kernel 0: one-time weight transpose  WkT[h][c][k] = (f16) kernels[h][k][c].
// 1 MB read / 0.5 MB write; runs once per launch, stays L2-resident.
// ---------------------------------------------------------------------------
__global__ __launch_bounds__(256) void gat_wtrans(
    const float* __restrict__ Wk, _Float16* __restrict__ WkT) {
  int o = blockIdx.x * 256 + threadIdx.x;   // ((h*FD + c)*F_IN + k)
  int k  = o & (F_IN - 1);
  int hc = o >> 9;
  int c  = hc & (FD - 1);
  int h  = hc >> 6;
  WkT[o] = (_Float16)Wk[((size_t)h * F_IN + k) * FD + c];
}

// ---------------------------------------------------------------------------
// Kernel 1: featsT[h,b,c,n] = (X[b,n,:] @ kernels[h,:,c])  transposed, f16.
// One workgroup (8 waves) per (h, b, 64-row tile). Pure global->WMMA:
// A-frag = 4x global_load_b128 (f32) + cvt_pk; B-frag = 2x global_load_b128.
// ---------------------------------------------------------------------------
__global__ __launch_bounds__(256) void gat_project(
    const float* __restrict__ X, const _Float16* __restrict__ WkT,
    _Float16* __restrict__ featsT) {
  int idx = blockIdx.x;
  int rb = idx & 3;               // 64-row tile
  int b  = (idx >> 2) & (B_ - 1);
  int h  = idx >> 9;
  int wave = threadIdx.x >> 5, lane = threadIdx.x & 31;
  int tm  = wave & 3;             // 16-row subtile
  int tnb = (wave >> 2) * 32;     // col strip base (0 or 32)
  int lm = lane & 15, hf = lane >> 4;

  v8f acc0 = {}; v8f acc1 = {};
  int n = rb * 64 + tm * 16 + lm;
  const float*    Xrow = X   + ((size_t)b * N_ + n) * F_IN;
  const _Float16* W0   = WkT + ((size_t)h * FD + tnb + lm) * F_IN;
  const _Float16* W1   = W0 + (size_t)16 * F_IN;

  for (int k0 = 0; k0 < F_IN; k0 += 32) {
    // A fragment: two 8-float runs of this lane's X row (32B-aligned)
    const float4* p0 = (const float4*)&Xrow[k0 + hf * 8];
    const float4* p1 = (const float4*)&Xrow[k0 + 16 + hf * 8];
    float4 a0 = p0[0], a1 = p0[1], a2 = p1[0], a3 = p1[1];
    v16h a;
    a[0]  = (_Float16)a0.x; a[1]  = (_Float16)a0.y;
    a[2]  = (_Float16)a0.z; a[3]  = (_Float16)a0.w;
    a[4]  = (_Float16)a1.x; a[5]  = (_Float16)a1.y;
    a[6]  = (_Float16)a1.z; a[7]  = (_Float16)a1.w;
    a[8]  = (_Float16)a2.x; a[9]  = (_Float16)a2.y;
    a[10] = (_Float16)a2.z; a[11] = (_Float16)a2.w;
    a[12] = (_Float16)a3.x; a[13] = (_Float16)a3.y;
    a[14] = (_Float16)a3.z; a[15] = (_Float16)a3.w;
    // B fragments: two 8-half runs in WkT (16B-aligned), same k order as A
    v16h bf0 = cat8(*(const v8h*)&W0[k0 + hf * 8],
                    *(const v8h*)&W0[k0 + 16 + hf * 8]);
    v16h bf1 = cat8(*(const v8h*)&W1[k0 + hf * 8],
                    *(const v8h*)&W1[k0 + 16 + hf * 8]);
    acc0 = __builtin_amdgcn_wmma_f32_16x16x32_f16(false, a, false, bf0,
                                                  (short)0, acc0, false, false);
    acc1 = __builtin_amdgcn_wmma_f32_16x16x32_f16(false, a, false, bf1,
                                                  (short)0, acc1, false, false);
  }

  // D-tile -> featsT[h,b][c][n]: 8 consecutive n per lane -> b128 stores
  int n0 = rb * 64 + tm * 16 + 8 * hf;
  size_t base = ((size_t)h * B_ + b) * FD;
  v8h h0, h1;
  for (int r = 0; r < 8; ++r) { h0[r] = (_Float16)acc0[r]; h1[r] = (_Float16)acc1[r]; }
  *(v8h*)&featsT[(base + tnb + lm) * N_ + n0]      = h0;
  *(v8h*)&featsT[(base + tnb + 16 + lm) * N_ + n0] = h1;
}

// ---------------------------------------------------------------------------
// Kernel 2: a_self/a_neigh[h,b,n] = featsT[h,b,:,n] . attn_vec
// Block (128 thr) per (h,b); thread = 2 nodes -> packed b32 loads.
// ---------------------------------------------------------------------------
__global__ __launch_bounds__(128) void gat_logits(
    const _Float16* __restrict__ featsT,
    const float* __restrict__ attn_self, const float* __restrict__ attn_neigh,
    float* __restrict__ a_self, float* __restrict__ a_neigh) {
  int hb = blockIdx.x;            // (h*B + b)
  int h = hb >> 7;                // B_ = 128
  int n0 = threadIdx.x * 2;
  const _Float16* f = featsT + (size_t)hb * FD * N_;
  float s1a = 0.f, s1b = 0.f, s2a = 0.f, s2b = 0.f;
  for (int c = 0; c < FD; ++c) {
    v2h v = *(const v2h*)&f[(size_t)c * N_ + n0];
    float va = (float)v[0], vb = (float)v[1];
    float ws = attn_self[h * FD + c], wn = attn_neigh[h * FD + c];
    s1a += va * ws; s1b += vb * ws;
    s2a += va * wn; s2b += vb * wn;
  }
  a_self[(size_t)hb * N_ + n0]      = s1a;
  a_self[(size_t)hb * N_ + n0 + 1]  = s1b;
  a_neigh[(size_t)hb * N_ + n0]     = s2a;
  a_neigh[(size_t)hb * N_ + n0 + 1] = s2b;
}

// ---------------------------------------------------------------------------
// Kernel 3: fused masked-softmax + node GEMM + head-mean view.
// One workgroup per (b, 16-row tile); wave w == head w. attn tile in LDS f16.
// Lane owns 8 CONTIGUOUS columns -> float4 mask loads, b128 attn stores.
// ---------------------------------------------------------------------------
__global__ __launch_bounds__(256) void gat_attention(
    const float* __restrict__ A, const float* __restrict__ a_self,
    const float* __restrict__ a_neigh, const _Float16* __restrict__ featsT,
    const float* __restrict__ biases, float* __restrict__ out,
    float* __restrict__ view) {
  __shared__ __align__(16) _Float16 attn_s[H_][16][N_];  // 64 KB
  int b = blockIdx.x >> 4;
  int row0 = (blockIdx.x & 15) * 16;
  int h = threadIdx.x >> 5;       // wave == head
  int lane = threadIdx.x & 31;
  int j8 = lane * 8;              // this lane's contiguous 8-column strip

  // ---- phase 1: per-row masked LeakyReLU softmax (f32, wave32 reductions)
  const float* anp = a_neigh + ((size_t)h * B_ + b) * N_;
  float4 an0 = ((const float4*)&anp[j8])[0];
  float4 an1 = ((const float4*)&anp[j8])[1];
  float an[8] = {an0.x, an0.y, an0.z, an0.w, an1.x, an1.y, an1.z, an1.w};
  const float* asp = a_self + ((size_t)h * B_ + b) * N_ + row0;
  for (int m = 0; m < 16; ++m) {
    float asv = asp[m];
    const float* Am = A + ((size_t)b * N_ + row0 + m) * N_ + j8;
    float4 m0 = ((const float4*)Am)[0];
    float4 m1 = ((const float4*)Am)[1];
    float msk[8] = {m0.x, m0.y, m0.z, m0.w, m1.x, m1.y, m1.z, m1.w};
    float e[8]; float mx = -3.0e38f;
    for (int jj = 0; jj < 8; ++jj) {
      float x = asv + an[jj];
      x = x > 0.f ? x : SLOPE * x;              // LeakyReLU(0.2)
      x += NEG_BIG * (1.0f - msk[jj]);          // mask non-edges
      e[jj] = x;
      mx = fmaxf(mx, x);
    }
    for (int off = 16; off > 0; off >>= 1) mx = fmaxf(mx, __shfl_xor(mx, off, 32));
    float sum = 0.f;
    for (int jj = 0; jj < 8; ++jj) { e[jj] = __expf(e[jj] - mx); sum += e[jj]; }
    for (int off = 16; off > 0; off >>= 1) sum += __shfl_xor(sum, off, 32);
    float inv = 1.0f / sum;
    v8h row;
    for (int jj = 0; jj < 8; ++jj) row[jj] = (_Float16)(e[jj] * inv);
    *(v8h*)&attn_s[h][m][j8] = row;             // single ds_store_b128
  }
  __syncthreads();

  // ---- phase 2: view = mean over heads; 16 contiguous cols per thread
  {
    int m  = threadIdx.x >> 4;          // 0..15
    int j0 = (threadIdx.x & 15) * 16;   // 0..240
    float accv[16];
    for (int e = 0; e < 16; ++e) accv[e] = 0.f;
    for (int hh = 0; hh < H_; ++hh) {
      v8h q0 = *(const v8h*)&attn_s[hh][m][j0];
      v8h q1 = *(const v8h*)&attn_s[hh][m][j0 + 8];
      for (int e = 0; e < 8; ++e) {
        accv[e]     += (float)q0[e];
        accv[8 + e] += (float)q1[e];
      }
    }
    float* vp = &view[((size_t)b * N_ + row0 + m) * N_ + j0];
    for (int q = 0; q < 4; ++q) {
      float4 w4 = make_float4(accv[4 * q] * 0.125f, accv[4 * q + 1] * 0.125f,
                              accv[4 * q + 2] * 0.125f, accv[4 * q + 3] * 0.125f);
      ((float4*)vp)[q] = w4;
    }
  }

  // ---- phase 3: node[16,64] = attn[16,256] @ feats[256,64], bias + relu
  int lm = lane & 15, hf = lane >> 4;
  const _Float16* fp = featsT + ((size_t)h * B_ + b) * FD * N_;  // [c][n]
  v8f acc[4] = {{}, {}, {}, {}};
  for (int k0 = 0; k0 < N_; k0 += 32) {
    // A fragment from LDS: two ds_load_b128
    v16h a = cat8(*(const v8h*)&attn_s[h][lm][k0 + hf * 8],
                  *(const v8h*)&attn_s[h][lm][k0 + 16 + hf * 8]);
    for (int nt = 0; nt < 4; ++nt) {   // A-fragment reused 4x
      int c = nt * 16 + lm;
      const _Float16* col = &fp[(size_t)c * N_ + k0 + hf * 8];
      v16h bf = cat8(*(const v8h*)col, *(const v8h*)(col + 16));  // b128 x2
      acc[nt] = __builtin_amdgcn_wmma_f32_16x16x32_f16(false, a, false, bf,
                                                       (short)0, acc[nt],
                                                       false, false);
    }
  }
  for (int nt = 0; nt < 4; ++nt) {
    int col = h * FD + nt * 16 + lm;
    float bias = biases[h * FD + nt * 16 + lm];
    for (int r = 0; r < 8; ++r) {
      int n = row0 + r + 8 * hf;
      float v = acc[nt][r] + bias;
      out[((size_t)b * N_ + n) * (H_ * FD) + col] = v > 0.f ? v : 0.f;  // relu
    }
  }
}

// ---------------------------------------------------------------------------
// Kernel 4: per-batch threshold + binarized map.
// sum(pooled @ alpha) == dot(colsum(pooled), rowsum(alpha))  -> 128-dot.
// ---------------------------------------------------------------------------
__global__ __launch_bounds__(256) void gat_threshold(
    const float* __restrict__ view, const float* __restrict__ alpha,
    float* __restrict__ dense4) {
  __shared__ float rs_alpha[128];
  __shared__ float red[256];
  __shared__ float thresh_s;
  int b = blockIdx.x, t = threadIdx.x;

  if (t < 128) {
    float s = 0.f;
    for (int k = 0; k < 128; ++k) s += alpha[t * 128 + k];
    rs_alpha[t] = s;
  }
  __syncthreads();

  float partial = 0.f;
  const float* vb = view + (size_t)b * N_ * N_;
  if (t < 128) {
    int j = t;
    float cs = 0.f;
    for (int i = 0; i < 128; ++i) {            // colsum of 2x2 maxpool
      const float* r0 = vb + (size_t)(2 * i) * N_ + 2 * j;
      const float* r1 = r0 + N_;
      cs += fmaxf(fmaxf(r0[0], r0[1]), fmaxf(r1[0], r1[1]));
    }
    partial = cs * rs_alpha[j];
  }
  red[t] = partial;
  __syncthreads();
  for (int off = 128; off > 0; off >>= 1) {
    if (t < off) red[t] += red[t + off];
    __syncthreads();
  }
  if (t == 0) thresh_s = 1.0f / (1.0f + __expf(-red[0]));  // sigmoid
  __syncthreads();

  float th = thresh_s;
  float* ob = dense4 + (size_t)b * N_ * N_;
  for (int it = 0; it < 256; ++it) {
    int idx = it * 256 + t;
    int i = idx >> 8, j = idx & 255;
    float v = vb[(size_t)i * N_ + j] + (i == j ? 1.0f : 0.0f);
    ob[(size_t)i * N_ + j] = (v < th) ? 1.0f : 0.0f;
  }
}

// ---------------------------------------------------------------------------
extern "C" void kernel_launch(void* const* d_in, const int* in_sizes, int n_in,
                              void* d_out, int out_size, void* d_ws, size_t ws_size,
                              hipStream_t stream) {
  const float* X          = (const float*)d_in[0];
  const float* A          = (const float*)d_in[1];
  const float* kernels    = (const float*)d_in[2];
  const float* biases     = (const float*)d_in[3];
  const float* attn_self  = (const float*)d_in[4];
  const float* attn_neigh = (const float*)d_in[5];
  const float* alpha      = (const float*)d_in[6];

  // workspace partition (every element is fully overwritten each call)
  char* ws = (char*)d_ws;
  _Float16* featsT = (_Float16*)ws;                               // 33,554,432 B
  float* a_self    = (float*)(ws + (size_t)33554432);             //  1,048,576 B
  float* a_neigh   = (float*)(ws + (size_t)34603008);             //  1,048,576 B
  float* view      = (float*)(ws + (size_t)35651584);             // 33,554,432 B
  _Float16* WkT    = (_Float16*)(ws + (size_t)69206016);          //    524,288 B

  float* out    = (float*)d_out;                   // [B,N,H*FD] = 16,777,216
  float* dense4 = out + (size_t)B_ * N_ * H_ * FD; // [B,N,N]    =  8,388,608

  hipLaunchKernelGGL(gat_wtrans,    dim3(H_ * FD * F_IN / 256), dim3(256), 0, stream,
                     kernels, WkT);
  hipLaunchKernelGGL(gat_project,   dim3(H_ * B_ * 4),    dim3(256), 0, stream,
                     X, WkT, featsT);
  hipLaunchKernelGGL(gat_logits,    dim3(H_ * B_),        dim3(128), 0, stream,
                     featsT, attn_self, attn_neigh, a_self, a_neigh);
  hipLaunchKernelGGL(gat_attention, dim3(B_ * (N_ / 16)), dim3(256), 0, stream,
                     A, a_self, a_neigh, featsT, biases, out, view);
  hipLaunchKernelGGL(gat_threshold, dim3(B_),             dim3(256), 0, stream,
                     view, alpha, dense4);
}
